// Quantizer_23244363006432
// MI455X (gfx1250) — matrix operand order repaired and checked by reading, not compiled, not run on hardware
//
#include <hip/hip_runtime.h>

// ---------------------------------------------------------------------------
// VQ-VAE nearest-codebook quantizer for MI455X (gfx1250, wave32, WMMA).
//   argmin_k ||z - w_k||^2  ==  argmax_k ( z.w_k - 0.5*||w_k||^2 )
// -0.5*||w||^2 is injected as the WMMA C-matrix, so the whole scoring is
// matrix-pipe work; the per-candidate update is a branchless select-argmax.
// Cross term via v_wmma_f32_16x16x32_f16 with z split into f16 hi + lo
// residual (error limited to w->f16 rounding; the output is an exact fp32
// gather of codebook rows, so only near-tie argmin flips are possible).
// All staging / output data movement is 16-byte vectorized.
// ---------------------------------------------------------------------------

typedef __attribute__((ext_vector_type(16))) _Float16 v16h;
typedef __attribute__((ext_vector_type(4)))  _Float16 v4h;
typedef __attribute__((ext_vector_type(8)))  float    v8f;
typedef __attribute__((ext_vector_type(4)))  float    v4f;
typedef __attribute__((ext_vector_type(4)))  unsigned v4u;
typedef __attribute__((ext_vector_type(4)))  int      v4i;

#define EMB        64     // embedding dim D
#define NCODES     512    // codebook size K
#define HW         4096   // 64*64 spatial
#define PIX        128    // pixels per workgroup
#define THREADS    256    // 8 waves
#define CB_STRIDE  72     // codebook LDS row stride in halves: 144B rows ->
                          // every A sub-row 16B aligned, banks conflict-free

#define CB_BYTES   (NCODES * CB_STRIDE * 2)   // 73728
#define ZT_BYTES   (EMB * PIX * 4)            // 32768
#define WSQ_BYTES  (NCODES * 4)               // 2048
#define BEST_BYTES (PIX * 4)                  // 512
#define SMEM_BYTES (CB_BYTES + ZT_BYTES + WSQ_BYTES + BEST_BYTES)  // 109056

__global__ __launch_bounds__(THREADS)
void vq_argmin_wmma(const float* __restrict__ z_e,
                    const float* __restrict__ weight,
                    float* __restrict__ out) {
    extern __shared__ char smem[];
    _Float16* cb16  = (_Float16*)smem;                               // [512][72] f16
    float*    ztile = (float*)(smem + CB_BYTES);                     // [64][128] f32
    float*    wsqn  = (float*)(smem + CB_BYTES + ZT_BYTES);          // [512] = -0.5*||w||^2
    int*      best  = (int*)(smem + CB_BYTES + ZT_BYTES + WSQ_BYTES);// [128]

    const int t  = threadIdx.x;
    const int g  = blockIdx.x;
    const int b  = g >> 5;               // image index (32 tiles per image)
    const int s0 = (g & 31) * PIX;       // spatial offset within image

    const float* zbase = z_e + (size_t)b * EMB * HW + s0;

    // ---- stage z tile, float4 (channel-major == B-matrix K x N layout) ----
    for (int i = t; i < EMB * PIX / 4; i += THREADS) {
        int c = i >> 5, p4 = (i & 31) * 4;
        *(v4f*)(ztile + c * PIX + p4) = *(const v4f*)(zbase + (size_t)c * HW + p4);
    }
    // ---- stage codebook as f16 (float4 load -> 8B packed-half store) ----
    for (int i = t; i < NCODES * EMB / 4; i += THREADS) {
        int k = i >> 4, f4 = (i & 15) * 4;
        v4f w = *(const v4f*)(weight + k * EMB + f4);
        v4h h = { (_Float16)w.x, (_Float16)w.y, (_Float16)w.z, (_Float16)w.w };
        *(v4h*)(cb16 + k * CB_STRIDE + f4) = h;
    }
    // ---- exact fp32 -0.5*||w_k||^2 from global (L2-resident), float4 ----
    for (int k = t; k < NCODES; k += THREADS) {
        const v4f* wr = (const v4f*)(weight + (size_t)k * EMB);
        v4f s = {};
        #pragma unroll
        for (int f = 0; f < EMB / 4; ++f) { v4f w = wr[f]; s += w * w; }
        wsqn[k] = -0.5f * (s.x + s.y + s.z + s.w);
    }
    __syncthreads();

    // ---- compute phase: wave w owns pixels [16w, 16w+16) ----
    const int wave = t >> 5;
    const int lane = t & 31;
    const int half = lane >> 4;       // lane group 0/1
    const int n    = lane & 15;       // N index (pixel within wave tile)
    const int p    = wave * 16 + n;

    // B fragments (z), loaded once, reused across all 32 code tiles.
    // B layout: lanes 0-15 hold K {0..7,16..23}, lanes 16-31 +8 (per chunk).
    v16h bhi[2], blo[2];
    #pragma unroll
    for (int ch = 0; ch < 2; ++ch) {
        #pragma unroll
        for (int i = 0; i < 16; ++i) {
            int f = ch * 32 + half * 8 + i + (i & 8);
            float zv = ztile[f * PIX + p];
            _Float16 h = (_Float16)zv;
            bhi[ch][i] = h;
            blo[ch][i] = (_Float16)(zv - (float)h);
        }
    }

    float bestv = -3.4e38f;   // argmax of (z.w - 0.5||w||^2)
    int   besti = 0;

    for (int tile = 0; tile < 32; ++tile) {
        const int tbase = tile * 16 + half * 8;   // codes in this lane's D rows

        // C matrix: D-layout VGPR r holds code (tbase + r) -> -0.5*||w||^2
        union { v8f v; v4f q[2]; } accu;
        accu.q[0] = *(const v4f*)(wsqn + tbase);
        accu.q[1] = *(const v4f*)(wsqn + tbase + 4);
        v8f acc = accu.v;

        const int code_a = tile * 16 + n;   // A-matrix row M = lane&15
        #pragma unroll
        for (int ch = 0; ch < 2; ++ch) {
            // A layout: lanes 0-15 K-seq {0..7,16..23}, lanes 16-31 +8.
            // 16B-aligned sub-rows -> two ds_load_b128 per chunk.
            union { v16h h; v4u q[2]; } a;
            const _Float16* arow = cb16 + code_a * CB_STRIDE + ch * 32 + half * 8;
            a.q[0] = *(const v4u*)(arow);        // K 0..7
            a.q[1] = *(const v4u*)(arow + 16);   // K 16..23
            acc = __builtin_amdgcn_wmma_f32_16x16x32_f16(
                      false, a.h, false, bhi[ch], (short)0, acc, false, false);
            acc = __builtin_amdgcn_wmma_f32_16x16x32_f16(
                      false, a.h, false, blo[ch], (short)0, acc, false, false);
        }
        // Branchless select-argmax; strict '>' keeps lowest code on ties
        // (codes visited in increasing order within a lane).
        #pragma unroll
        for (int r = 0; r < 8; ++r) {
            float d = acc[r];
            int code = tbase + r;
            bool c2 = d > bestv;
            besti = c2 ? code : besti;
            bestv = c2 ? d : bestv;
        }
    }

    // merge the two lane halves covering complementary codes of pixel p
    float ov = __shfl_xor(bestv, 16, 32);
    int   oi = __shfl_xor(besti, 16, 32);
    bool take = (ov > bestv) || (ov == bestv && oi < besti);
    besti = take ? oi : besti;
    if (half == 0) best[p] = besti;
    __syncthreads();

    // ---- output: exact fp32 codebook gather, 16B coalesced NCHW stores ----
    float* obase = out + (size_t)b * EMB * HW + s0;
    for (int i = t; i < EMB * PIX / 4; i += THREADS) {
        int c = i >> 5, p4 = (i & 31) * 4;
        v4i k4 = *(const v4i*)(best + p4);
        v4f o = { weight[(size_t)k4.x * EMB + c],
                  weight[(size_t)k4.y * EMB + c],
                  weight[(size_t)k4.z * EMB + c],
                  weight[(size_t)k4.w * EMB + c] };
        *(v4f*)(obase + (size_t)c * HW + p4) = o;
    }
}

extern "C" void kernel_launch(void* const* d_in, const int* in_sizes, int n_in,
                              void* d_out, int out_size, void* d_ws, size_t ws_size,
                              hipStream_t stream) {
    const float* z_e    = (const float*)d_in[0];
    const float* weight = (const float*)d_in[1];
    float*       out    = (float*)d_out;

    (void)d_ws; (void)ws_size; (void)n_in; (void)out_size;

    // allow >64KB dynamic LDS (CDNA5 WGP has 320KB)
    hipFuncSetAttribute(reinterpret_cast<const void*>(vq_argmin_wmma),
                        hipFuncAttributeMaxDynamicSharedMemorySize, SMEM_BYTES);

    const int npix    = in_sizes[0] / EMB;   // 131072
    const int nblocks = npix / PIX;          // 1024

    vq_argmin_wmma<<<nblocks, THREADS, SMEM_BYTES, stream>>>(z_e, weight, out);
}
